// Decoder_32427003085175
// MI455X (gfx1250) — compile-verified, matrix operand order checked by reading, not run on hardware
//
#include <hip/hip_runtime.h>
#include <hip/hip_bf16.h>
#include <math.h>

#define Hh 512
#define Vv 32000
#define Bb 32
#define Ss 128
#define Tt 48
#define BOS_TOK 1

typedef __attribute__((ext_vector_type(16))) __bf16 v16bf;
typedef __attribute__((ext_vector_type(8)))  float  v8f;

// ---------------- weight/activation conversion f32 -> bf16 -----------------------
__global__ void cvt_f32_bf16(const float* __restrict__ in, __bf16* __restrict__ out, int n) {
    int i = blockIdx.x * 256 + threadIdx.x;
    if (i < n) out[i] = (__bf16)in[i];
}

__global__ void add_bias2(const float* __restrict__ a, const float* __restrict__ b,
                          float* __restrict__ o, int n) {
    int i = blockIdx.x * 256 + threadIdx.x;
    if (i < n) o[i] = a[i] + b[i];
}

// ---------------- WMMA GEMM: C[m,n] (+)= sum_k A[m,k]*W[n,k] + bias[n] -----------
// A: bf16 row-major (lda), W: bf16 row-major (N x K), C: f32 (ldc) or Cb: bf16.
// REQUIRES N % 64 == 0 and K % 64 == 0 (true at all call sites).
// Block = 256 thr = 8 waves; each wave computes a 16x64 strip (4 N tiles).
// k-loop is software-pipelined with two fragment groups (double buffering) so the
// next group's 10 b128 loads overlap the current group's 4 WMMAs.
__global__ void wmma_gemm_bf16(const __bf16* __restrict__ A, int lda,
                               const __bf16* __restrict__ W,
                               const float* __restrict__ bias,
                               float* __restrict__ C, __bf16* __restrict__ Cb,
                               long long ldc, int N, int K, int accFlag, int actFlag) {
    const int lane = threadIdx.x & 31;
    const int wave = threadIdx.x >> 5;
    const int ngrp = N >> 6;                       // 64-column groups
    const int grp  = blockIdx.x * 8 + wave;
    if (grp >= ngrp) return;                       // wave-uniform tail guard
    const int tn0  = grp * 4;                      // first 16-col tile of this wave
    const int tm   = blockIdx.y;

    const int arow  = tm * 16 + (lane & 15);       // A row for this lane
    const int koff  = (lane >> 4) * 16;            // K sub-range per lane half
    const int cl    = lane & 15;                   // C column within tile
    const int mbase = tm * 16 + ((lane >> 4) << 3);

    const int cn0 = (tn0 + 0) * 16 + cl;
    const int cn1 = (tn0 + 1) * 16 + cl;
    const int cn2 = (tn0 + 2) * 16 + cl;
    const int cn3 = (tn0 + 3) * 16 + cl;

    v8f acc0 = {}, acc1 = {}, acc2 = {}, acc3 = {};
    if (accFlag) {
#pragma unroll
        for (int i = 0; i < 8; ++i) {
            const long long r = (long long)(mbase + i) * ldc;
            acc0[i] = C[r + cn0];
            acc1[i] = C[r + cn1];
            acc2[i] = C[r + cn2];
            acc3[i] = C[r + cn3];
        }
    }

    const __bf16* ap = A + (long long)arow * lda + koff;
    const __bf16* wp = W + ((long long)tn0 * 16 + cl) * K + koff;
    const long long wt = (long long)16 * K;        // stride between N tiles in W

    // prologue: group X at k=0
    v16bf ax = *(const v16bf*)(ap);
    v16bf x0 = *(const v16bf*)(wp);
    v16bf x1 = *(const v16bf*)(wp + wt);
    v16bf x2 = *(const v16bf*)(wp + 2 * wt);
    v16bf x3 = *(const v16bf*)(wp + 3 * wt);

    int k = 0;
    for (; k < K - 64; k += 64) {
        // issue group Y loads (k+32) before consuming group X
        v16bf ay = *(const v16bf*)(ap + k + 32);
        v16bf y0 = *(const v16bf*)(wp + k + 32);
        v16bf y1 = *(const v16bf*)(wp + wt + k + 32);
        v16bf y2 = *(const v16bf*)(wp + 2 * wt + k + 32);
        v16bf y3 = *(const v16bf*)(wp + 3 * wt + k + 32);
        acc0 = __builtin_amdgcn_wmma_f32_16x16x32_bf16(false, ax, false, x0, (short)0, acc0, false, false);
        acc1 = __builtin_amdgcn_wmma_f32_16x16x32_bf16(false, ax, false, x1, (short)0, acc1, false, false);
        acc2 = __builtin_amdgcn_wmma_f32_16x16x32_bf16(false, ax, false, x2, (short)0, acc2, false, false);
        acc3 = __builtin_amdgcn_wmma_f32_16x16x32_bf16(false, ax, false, x3, (short)0, acc3, false, false);
        // issue group X loads (k+64) before consuming group Y
        ax = *(const v16bf*)(ap + k + 64);
        x0 = *(const v16bf*)(wp + k + 64);
        x1 = *(const v16bf*)(wp + wt + k + 64);
        x2 = *(const v16bf*)(wp + 2 * wt + k + 64);
        x3 = *(const v16bf*)(wp + 3 * wt + k + 64);
        acc0 = __builtin_amdgcn_wmma_f32_16x16x32_bf16(false, ay, false, y0, (short)0, acc0, false, false);
        acc1 = __builtin_amdgcn_wmma_f32_16x16x32_bf16(false, ay, false, y1, (short)0, acc1, false, false);
        acc2 = __builtin_amdgcn_wmma_f32_16x16x32_bf16(false, ay, false, y2, (short)0, acc2, false, false);
        acc3 = __builtin_amdgcn_wmma_f32_16x16x32_bf16(false, ay, false, y3, (short)0, acc3, false, false);
    }
    // epilogue: k = K-64 (group X already loaded), then K-32
    {
        v16bf ay = *(const v16bf*)(ap + k + 32);
        v16bf y0 = *(const v16bf*)(wp + k + 32);
        v16bf y1 = *(const v16bf*)(wp + wt + k + 32);
        v16bf y2 = *(const v16bf*)(wp + 2 * wt + k + 32);
        v16bf y3 = *(const v16bf*)(wp + 3 * wt + k + 32);
        acc0 = __builtin_amdgcn_wmma_f32_16x16x32_bf16(false, ax, false, x0, (short)0, acc0, false, false);
        acc1 = __builtin_amdgcn_wmma_f32_16x16x32_bf16(false, ax, false, x1, (short)0, acc1, false, false);
        acc2 = __builtin_amdgcn_wmma_f32_16x16x32_bf16(false, ax, false, x2, (short)0, acc2, false, false);
        acc3 = __builtin_amdgcn_wmma_f32_16x16x32_bf16(false, ax, false, x3, (short)0, acc3, false, false);
        acc0 = __builtin_amdgcn_wmma_f32_16x16x32_bf16(false, ay, false, y0, (short)0, acc0, false, false);
        acc1 = __builtin_amdgcn_wmma_f32_16x16x32_bf16(false, ay, false, y1, (short)0, acc1, false, false);
        acc2 = __builtin_amdgcn_wmma_f32_16x16x32_bf16(false, ay, false, y2, (short)0, acc2, false, false);
        acc3 = __builtin_amdgcn_wmma_f32_16x16x32_bf16(false, ay, false, y3, (short)0, acc3, false, false);
    }

    const float bn0 = bias ? bias[cn0] : 0.0f;
    const float bn1 = bias ? bias[cn1] : 0.0f;
    const float bn2 = bias ? bias[cn2] : 0.0f;
    const float bn3 = bias ? bias[cn3] : 0.0f;
#pragma unroll
    for (int i = 0; i < 8; ++i) {
        float v0 = acc0[i] + bn0, v1 = acc1[i] + bn1;
        float v2 = acc2[i] + bn2, v3 = acc3[i] + bn3;
        if (actFlag) { v0 = tanhf(v0); v1 = tanhf(v1); v2 = tanhf(v2); v3 = tanhf(v3); }
        const long long r = (long long)(mbase + i) * ldc;
        if (Cb) {
            Cb[r + cn0] = (__bf16)v0; Cb[r + cn1] = (__bf16)v1;
            Cb[r + cn2] = (__bf16)v2; Cb[r + cn3] = (__bf16)v3;
        } else {
            C[r + cn0] = v0; C[r + cn1] = v1; C[r + cn2] = v2; C[r + cn3] = v3;
        }
    }
}

// ---------------- attention score: e[b,s] = sum_h tanh(WhE[b,s,h]+proj[b,h])*va[h]
__global__ void attend_e_k(const float* __restrict__ WhE, const float* __restrict__ proj,
                           const float* __restrict__ va, float* __restrict__ e) {
    __shared__ float red[256];
    const int bs = blockIdx.x;            // b*S + s
    const int b  = bs / Ss;
    float p = 0.0f;
    for (int h = threadIdx.x; h < Hh; h += 256)
        p += tanhf(WhE[(long long)bs * Hh + h] + proj[b * Hh + h]) * va[h];
    red[threadIdx.x] = p; __syncthreads();
    for (int st = 128; st > 0; st >>= 1) {
        if (threadIdx.x < st) red[threadIdx.x] += red[threadIdx.x + st];
        __syncthreads();
    }
    if (threadIdx.x == 0) e[bs] = red[0];
}

// ---------------- softmax over S (one block per batch, 128 threads) --------------
__global__ void softmax_s_k(const float* __restrict__ e, float* __restrict__ a) {
    __shared__ float red[128];
    __shared__ float bc;
    const int b = blockIdx.x, s = threadIdx.x;
    float v = e[b * Ss + s];
    red[s] = v; __syncthreads();
    for (int st = 64; st > 0; st >>= 1) {
        if (s < st) red[s] = fmaxf(red[s], red[s + st]);
        __syncthreads();
    }
    if (s == 0) bc = red[0];
    __syncthreads();
    float ex = expf(v - bc);
    red[s] = ex; __syncthreads();
    for (int st = 64; st > 0; st >>= 1) {
        if (s < st) red[s] += red[s + st];
        __syncthreads();
    }
    if (s == 0) bc = red[0];
    __syncthreads();
    a[b * Ss + s] = ex / bc;
}

// ---------------- context: c[b,h] = sum_s a[b,s]*enc[b,s,h]; f32 + bf16 outs -----
__global__ void context_k(const float* __restrict__ a, const float* __restrict__ enc,
                          float* __restrict__ c, __bf16* __restrict__ cb) {
    const int b = blockIdx.x, t = threadIdx.x;   // 256 threads, 2 h each
    float s0 = 0.0f, s1 = 0.0f;
    for (int s = 0; s < Ss; ++s) {
        const float w = a[b * Ss + s];
        const float* er = enc + (long long)(b * Ss + s) * Hh;
        s0 += w * er[t];
        s1 += w * er[t + 256];
    }
    c[b * Hh + t]        = s0;
    c[b * Hh + t + 256]  = s1;
    cb[b * Hh + t]       = (__bf16)s0;
    cb[b * Hh + t + 256] = (__bf16)s1;
}

// ---------------- gi = [emb[tok_t], c]  (bf16, feeds Wih GEMM) -------------------
__global__ void build_gi_k(const int* __restrict__ target, const float* __restrict__ emb,
                           const float* __restrict__ c, __bf16* __restrict__ gib, int t) {
    const int i = blockIdx.x * 256 + threadIdx.x;
    if (i >= Bb * Hh) return;
    const int b = i / Hh, h = i % Hh;
    const int tok = (t == 0) ? BOS_TOK : target[b * Tt + t - 1];
    gib[(long long)b * 2 * Hh + h]      = (__bf16)emb[(long long)tok * Hh + h];
    gib[(long long)b * 2 * Hh + Hh + h] = (__bf16)c[i];
}

// ---------------- GRU update; emits h_new (f32+bf16) and cat=[h_new,c] (bf16) ----
__global__ void gru_k(const float* __restrict__ gx, const float* __restrict__ gh,
                      const float* __restrict__ hprev, const float* __restrict__ c,
                      float* __restrict__ hnew, __bf16* __restrict__ hnewb,
                      __bf16* __restrict__ catb) {
    const int i = blockIdx.x * 256 + threadIdx.x;
    if (i >= Bb * Hh) return;
    const int b = i / Hh, h = i % Hh;
    const float* gxr = gx + (long long)b * 3 * Hh;
    const float* ghr = gh + (long long)b * 3 * Hh;
    const float rg = 1.0f / (1.0f + expf(-(gxr[h] + ghr[h])));
    const float zg = 1.0f / (1.0f + expf(-(gxr[Hh + h] + ghr[Hh + h])));
    const float ng = tanhf(gxr[2 * Hh + h] + rg * ghr[2 * Hh + h]);
    const float hv = (1.0f - zg) * ng + zg * hprev[i];
    hnew[i]  = hv;
    hnewb[i] = (__bf16)hv;
    catb[(long long)b * 2 * Hh + h]      = (__bf16)hv;
    catb[(long long)b * 2 * Hh + Hh + h] = (__bf16)c[i];
}

// ---------------- softmax over V in-place on d_out row (b, t) --------------------
__global__ void softmax_v_k(float* __restrict__ out, int t) {
    __shared__ float red[256];
    __shared__ float bc;
    const int b = blockIdx.x;
    float* row = out + ((long long)b * Tt + t) * (long long)Vv;
    float m = -1e30f;
    for (int v = threadIdx.x; v < Vv; v += 256) m = fmaxf(m, row[v]);
    red[threadIdx.x] = m; __syncthreads();
    for (int st = 128; st > 0; st >>= 1) {
        if (threadIdx.x < st) red[threadIdx.x] = fmaxf(red[threadIdx.x], red[threadIdx.x + st]);
        __syncthreads();
    }
    if (threadIdx.x == 0) bc = red[0];
    __syncthreads();
    m = bc;
    float s = 0.0f;
    for (int v = threadIdx.x; v < Vv; v += 256) {
        float ex = expf(row[v] - m);
        row[v] = ex;
        s += ex;
    }
    __syncthreads();
    red[threadIdx.x] = s; __syncthreads();
    for (int st = 128; st > 0; st >>= 1) {
        if (threadIdx.x < st) red[threadIdx.x] += red[threadIdx.x + st];
        __syncthreads();
    }
    if (threadIdx.x == 0) bc = red[0];
    __syncthreads();
    const float inv = 1.0f / bc;
    for (int v = threadIdx.x; v < Vv; v += 256) row[v] *= inv;
}

// ---------------- p_gen = sigmoid(sum_h tanh(sbuf+pba)*pva) ----------------------
__global__ void pgen_k(const float* __restrict__ sbuf, const float* __restrict__ pba,
                       const float* __restrict__ pva, float* __restrict__ pgen) {
    __shared__ float red[256];
    const int b = blockIdx.x;
    float p = 0.0f;
    for (int h = threadIdx.x; h < Hh; h += 256)
        p += tanhf(sbuf[b * Hh + h] + pba[h]) * pva[h];
    red[threadIdx.x] = p; __syncthreads();
    for (int st = 128; st > 0; st >>= 1) {
        if (threadIdx.x < st) red[threadIdx.x] += red[threadIdx.x + st];
        __syncthreads();
    }
    if (threadIdx.x == 0) pgen[b] = 1.0f / (1.0f + expf(-red[0]));
}

// ---------------- P_final = p_gen * P_vocab (in-place) ---------------------------
__global__ void scale_k(float* __restrict__ out, const float* __restrict__ pgen, int t) {
    const long long i = (long long)blockIdx.x * 256 + threadIdx.x;
    if (i >= (long long)Bb * Vv) return;
    const int b = (int)(i / Vv);
    const int v = (int)(i % Vv);
    out[((long long)b * Tt + t) * (long long)Vv + v] *= pgen[b];
}

// ---------------- scatter add of copy distribution -------------------------------
__global__ void scatter_k(float* __restrict__ out, const int* __restrict__ src_tokens,
                          const float* __restrict__ a, const float* __restrict__ pgen, int t) {
    const int i = blockIdx.x * 256 + threadIdx.x;
    if (i >= Bb * Ss) return;
    const int b = i / Ss;
    atomicAdd(&out[((long long)b * Tt + t) * (long long)Vv + src_tokens[i]],
              (1.0f - pgen[b]) * a[i]);
}

// ---------------- final extras: h_last, a_last, pgen_last ------------------------
__global__ void finalize_k(float* __restrict__ out, const float* __restrict__ hlast,
                           const float* __restrict__ a, const float* __restrict__ pgen) {
    const long long base = (long long)Bb * Tt * Vv;
    const int i = blockIdx.x * 256 + threadIdx.x;
    if (i < Bb * Hh) out[base + i] = hlast[i];
    if (i < Bb * Ss) out[base + Bb * Hh + i] = a[i];
    if (i < Bb)      out[base + Bb * Hh + Bb * Ss + i] = pgen[i];
}

extern "C" void kernel_launch(void* const* d_in, const int* in_sizes, int n_in,
                              void* d_out, int out_size, void* d_ws, size_t ws_size,
                              hipStream_t stream) {
    (void)in_sizes; (void)n_in; (void)out_size; (void)ws_size;
    const int*   target = (const int*)d_in[0];
    const float* hidden = (const float*)d_in[1];
    const float* enc    = (const float*)d_in[2];
    const int*   srctok = (const int*)d_in[3];
    const float* emb    = (const float*)d_in[4];
    const float* Wih    = (const float*)d_in[5];
    const float* Whh    = (const float*)d_in[6];
    const float* bih    = (const float*)d_in[7];
    const float* bhh    = (const float*)d_in[8];
    const float* aWh    = (const float*)d_in[9];
    const float* abh    = (const float*)d_in[10];
    const float* aWs    = (const float*)d_in[11];
    const float* abs_   = (const float*)d_in[12];
    const float* ava    = (const float*)d_in[13];
    const float* W1     = (const float*)d_in[14];
    const float* b1     = (const float*)d_in[15];
    const float* W2     = (const float*)d_in[16];
    const float* bW2    = (const float*)d_in[17];
    const float* b2     = (const float*)d_in[18];
    const float* pWsh   = (const float*)d_in[19];
    const float* pbsh   = (const float*)d_in[20];
    const float* pWse   = (const float*)d_in[21];
    const float* pbse   = (const float*)d_in[22];
    const float* pWsc   = (const float*)d_in[23];
    const float* pbsc   = (const float*)d_in[24];
    const float* pva    = (const float*)d_in[25];
    const float* pba    = (const float*)d_in[26];
    const float* paWh   = (const float*)d_in[27];
    const float* pabh   = (const float*)d_in[28];
    const float* paWs   = (const float*)d_in[29];
    const float* pabs   = (const float*)d_in[30];
    const float* pava   = (const float*)d_in[31];
    float* out = (float*)d_out;

    // ---- workspace carve-up (256B aligned) ----
    char* base = (char*)d_ws;
    size_t off = 0;
    auto alloc = [&](size_t bytes) -> char* {
        char* p = base + off;
        off = (off + bytes + 255) & ~(size_t)255;
        return p;
    };
    float*  WhE   = (float*)alloc((size_t)Bb * Ss * Hh * 4);
    float*  pWhE  = (float*)alloc((size_t)Bb * Ss * Hh * 4);
    float*  proj  = (float*)alloc((size_t)Bb * Hh * 4);
    float*  e     = (float*)alloc((size_t)Bb * Ss * 4);
    float*  a     = (float*)alloc((size_t)Bb * Ss * 4);
    float*  a2    = (float*)alloc((size_t)Bb * Ss * 4);
    float*  c     = (float*)alloc((size_t)Bb * Hh * 4);
    float*  c2    = (float*)alloc((size_t)Bb * Hh * 4);
    float*  gx    = (float*)alloc((size_t)Bb * 3 * Hh * 4);
    float*  gh    = (float*)alloc((size_t)Bb * 3 * Hh * 4);
    float*  hf0   = (float*)alloc((size_t)Bb * Hh * 4);
    float*  hf1   = (float*)alloc((size_t)Bb * Hh * 4);
    float*  t1f   = (float*)alloc((size_t)Bb * Hh * 4);   // dummy f32 (unused store path)
    float*  sbuf  = (float*)alloc((size_t)Bb * Hh * 4);
    float*  pgen  = (float*)alloc((size_t)Bb * 4);
    float*  cbias = (float*)alloc((size_t)Vv * 4);
    // bf16 activations
    __bf16* encb  = (__bf16*)alloc((size_t)Bb * Ss * Hh * 2);
    __bf16* gib   = (__bf16*)alloc((size_t)Bb * 2 * Hh * 2);
    __bf16* catb  = (__bf16*)alloc((size_t)Bb * 2 * Hh * 2);
    __bf16* hbi   = (__bf16*)alloc((size_t)Bb * Hh * 2);
    __bf16* hb0   = (__bf16*)alloc((size_t)Bb * Hh * 2);
    __bf16* hb1   = (__bf16*)alloc((size_t)Bb * Hh * 2);
    __bf16* t1b   = (__bf16*)alloc((size_t)Bb * Hh * 2);
    __bf16* cb    = (__bf16*)alloc((size_t)Bb * Hh * 2);
    __bf16* c2b   = (__bf16*)alloc((size_t)Bb * Hh * 2);
    // bf16 weights
    __bf16* W2b   = (__bf16*)alloc((size_t)Vv * Hh * 2);
    __bf16* Wihb  = (__bf16*)alloc((size_t)3 * Hh * 2 * Hh * 2);
    __bf16* Whhb  = (__bf16*)alloc((size_t)3 * Hh * Hh * 2);
    __bf16* W1b   = (__bf16*)alloc((size_t)Hh * 2 * Hh * 2);
    __bf16* aWhb  = (__bf16*)alloc((size_t)Hh * Hh * 2);
    __bf16* aWsb  = (__bf16*)alloc((size_t)Hh * Hh * 2);
    __bf16* paWhb = (__bf16*)alloc((size_t)Hh * Hh * 2);
    __bf16* paWsb = (__bf16*)alloc((size_t)Hh * Hh * 2);
    __bf16* pWshb = (__bf16*)alloc((size_t)Hh * Hh * 2);
    __bf16* pWseb = (__bf16*)alloc((size_t)Hh * Hh * 2);
    __bf16* pWscb = (__bf16*)alloc((size_t)Hh * Hh * 2);

    auto cvt = [&](const float* src, __bf16* dst, int n) {
        cvt_f32_bf16<<<(n + 255) / 256, 256, 0, stream>>>(src, dst, n);
    };
    // C = f32 store, Cb = bf16 store (if non-null)
    auto gemm = [&](const __bf16* A, int lda, const __bf16* Wt, const float* bias,
                    float* C, __bf16* Cb, long long ldc, int M, int N, int K,
                    int acc, int act) {
        dim3 g(((N >> 6) + 7) / 8, M / 16);   // 8 waves x 4 N-tiles (64 cols) per block
        wmma_gemm_bf16<<<g, 256, 0, stream>>>(A, lda, Wt, bias, C, Cb, ldc, N, K, acc, act);
    };

    // ---- one-time: bf16 weights/activations, combined vocab bias, enc projections
    cvt(W2,   W2b,   Vv * Hh);
    cvt(Wih,  Wihb,  3 * Hh * 2 * Hh);
    cvt(Whh,  Whhb,  3 * Hh * Hh);
    cvt(W1,   W1b,   Hh * 2 * Hh);
    cvt(aWh,  aWhb,  Hh * Hh);
    cvt(aWs,  aWsb,  Hh * Hh);
    cvt(paWh, paWhb, Hh * Hh);
    cvt(paWs, paWsb, Hh * Hh);
    cvt(pWsh, pWshb, Hh * Hh);
    cvt(pWse, pWseb, Hh * Hh);
    cvt(pWsc, pWscb, Hh * Hh);
    cvt(enc,    encb, Bb * Ss * Hh);
    cvt(hidden, hbi,  Bb * Hh);
    add_bias2<<<(Vv + 255) / 256, 256, 0, stream>>>(bW2, b2, cbias, Vv);
    gemm(encb, Hh, aWhb,  abh,  WhE,  nullptr, Hh, Bb * Ss, Hh, Hh, 0, 0);
    gemm(encb, Hh, paWhb, pabh, pWhE, nullptr, Hh, Bb * Ss, Hh, Hh, 0, 0);

    // ---- time loop (teacher forcing) ----
    for (int t = 0; t < Tt; ++t) {
        const __bf16* hcb = (t == 0) ? hbi    : ((t & 1) ? hb0 : hb1);
        const float*  hcf = (t == 0) ? hidden : ((t & 1) ? hf0 : hf1);
        __bf16* hbN = (t & 1) ? hb1 : hb0;
        float*  hfN = (t & 1) ? hf1 : hf0;
        // decoder attention
        gemm(hcb, Hh, aWsb, abs_, proj, nullptr, Hh, Bb, Hh, Hh, 0, 0);
        attend_e_k<<<Bb * Ss, 256, 0, stream>>>(WhE, proj, ava, e);
        softmax_s_k<<<Bb, 128, 0, stream>>>(e, a);
        context_k<<<Bb, 256, 0, stream>>>(a, enc, c, cb);
        // GRU
        build_gi_k<<<(Bb * Hh + 255) / 256, 256, 0, stream>>>(target, emb, c, gib, t);
        gemm(gib, 2 * Hh, Wihb, bih, gx, nullptr, 3 * Hh, Bb, 3 * Hh, 2 * Hh, 0, 0);
        gemm(hcb, Hh,     Whhb, bhh, gh, nullptr, 3 * Hh, Bb, 3 * Hh, Hh,     0, 0);
        gru_k<<<(Bb * Hh + 255) / 256, 256, 0, stream>>>(gx, gh, hcf, c, hfN, hbN, catb);
        // vocab distribution -> directly into d_out rows (b, t)
        gemm(catb, 2 * Hh, W1b, b1, t1f, t1b, Hh, Bb, Hh, 2 * Hh, 0, 1 /*tanh, bf16 out*/);
        gemm(t1b, Hh, W2b, cbias, out + (long long)t * Vv, nullptr, (long long)Tt * Vv,
             Bb, Vv, Hh, 0, 0);
        softmax_v_k<<<Bb, 256, 0, stream>>>(out, t);
        // pointer attention + gate
        gemm(hbN, Hh, paWsb, pabs, proj, nullptr, Hh, Bb, Hh, Hh, 0, 0);
        attend_e_k<<<Bb * Ss, 256, 0, stream>>>(pWhE, proj, pava, e);
        softmax_s_k<<<Bb, 128, 0, stream>>>(e, a2);
        context_k<<<Bb, 256, 0, stream>>>(a2, enc, c2, c2b);
        gemm(hbN, Hh, pWshb, pbsh, sbuf, nullptr, Hh, Bb, Hh, Hh, 0, 0);
        gemm(hbN, Hh, pWseb, pbse, sbuf, nullptr, Hh, Bb, Hh, Hh, 1, 0);
        gemm(c2b, Hh, pWscb, pbsc, sbuf, nullptr, Hh, Bb, Hh, Hh, 1, 0);
        pgen_k<<<Bb, 256, 0, stream>>>(sbuf, pba, pva, pgen);
        // combine: P_final = p_gen*P_vocab + scatter((1-p_gen)*a)
        scale_k<<<(int)(((long long)Bb * Vv + 255) / 256), 256, 0, stream>>>(out, pgen, t);
        scatter_k<<<(Bb * Ss + 255) / 256, 256, 0, stream>>>(out, srctok, a, pgen, t);
        if (t == Tt - 1)
            finalize_k<<<(Bb * Hh + 255) / 256, 256, 0, stream>>>(out, hfN, a, pgen);
    }
}
